// UncondHandwritingGenerator_30107720745089
// MI455X (gfx1250) — compile-verified, matrix operand order checked by reading, not run on hardware
//
#include <hip/hip_runtime.h>
#include <hip/hip_bf16.h>

// ---------------------------------------------------------------------------
// Problem constants
// ---------------------------------------------------------------------------
#define SEQ   16384
#define DIN   3
#define HID   400
#define G4H   1600          // 4*HID
#define KP    416           // HID padded to multiple of 32 for WMMA K
#define OUTC  121           // 1 + 6*20
#define SM1   (SEQ - 1)     // 16383 rows kept in output

// recurrence partitioning
#define REC_NWG      10
#define REC_UPW      40      // hidden units per workgroup
#define REC_ROWS     160     // 4 gates * REC_UPW
#define REC_THREADS  320
#define REC_WPITCH   404     // padded row pitch (floats) for LDS weight slice

// LDS bytes for recurrence kernel (floats): W slice + h + pre + c
#define REC_SMEM_FLOATS (REC_ROWS * REC_WPITCH + 404 + REC_ROWS + REC_UPW)
#define REC_SMEM_BYTES  (REC_SMEM_FLOATS * 4)

typedef __attribute__((ext_vector_type(16))) _Float16 v16h;
typedef __attribute__((ext_vector_type(8)))  _Float16 v8h;
typedef __attribute__((ext_vector_type(8)))  float    v8f;

// ---------------------------------------------------------------------------
// Small dense projection from the 3-dim input: out[s,o] = b0[o](+b1[o]) + x[s,:3]·W[o,:3]
// Used for xg of layer 1 (O=1600, b0=bih, b1=bhh) and the two skip_in projections.
// ---------------------------------------------------------------------------
__global__ void proj3_kernel(const float* __restrict__ x, const float* __restrict__ W,
                             const float* __restrict__ b0, const float* __restrict__ b1,
                             float* __restrict__ out, int O) {
    size_t idx = (size_t)blockIdx.x * blockDim.x + threadIdx.x;
    size_t total = (size_t)SEQ * (size_t)O;
    if (idx >= total) return;
    int    o = (int)(idx % O);
    size_t s = idx / O;
    float acc = b0[o] + (b1 ? b1[o] : 0.0f);
    const float* xr = x + s * 3;
    const float* wr = W + (size_t)o * 3;
    acc += xr[0] * wr[0] + xr[1] * wr[1] + xr[2] * wr[2];
    out[idx] = acc;
}

// ---------------------------------------------------------------------------
// fp32 -> fp16 conversion with optional elementwise add and K-padding to Kp.
// ---------------------------------------------------------------------------
__global__ void cvt_f16_pad_kernel(const float* __restrict__ A, const float* __restrict__ Badd,
                                   _Float16* __restrict__ out, int M, int K, int Kp) {
    size_t idx = (size_t)blockIdx.x * blockDim.x + threadIdx.x;
    size_t total = (size_t)M * (size_t)Kp;
    if (idx >= total) return;
    int    k = (int)(idx % Kp);
    size_t m = idx / Kp;
    float v = 0.0f;
    if (k < K) {
        v = A[m * K + k];
        if (Badd) v += Badd[m * K + k];
    }
    out[idx] = (_Float16)v;
}

// ---------------------------------------------------------------------------
// WMMA fragment loader: 16x32 f16 tile, A/B symmetric.
// Row-major source with pitch `pitch` (halves). Lane mapping per CDNA5 ISA:
//   lanes 0-15 : row = lane,    K = {kb..kb+7, kb+16..kb+23} with kb = k0
//   lanes 16-31: row = lane-16, K = same with kb = k0 + 8
// ---------------------------------------------------------------------------
__device__ __forceinline__ v16h load_frag16x32(const _Float16* __restrict__ base,
                                               int pitch, int row0, int k0) {
    int lane = threadIdx.x & 31;
    int r  = row0 + (lane & 15);
    int kb = k0 + ((lane & 16) ? 8 : 0);
    const _Float16* p = base + (size_t)r * pitch + kb;
    union { v16h v; struct { v8h lo; v8h hi; } h; } u;
    u.h.lo = *(const v8h*)(p);
    u.h.hi = *(const v8h*)(p + 16);
    return u.v;
}

// ---------------------------------------------------------------------------
// WMMA GEMM: C[M,N] (fp32) = A[M,Kp](f16) * B[N,Kp](f16)^T + bias0 + bias1 (+C)
// Register-blocked: each wave owns a 64x16 macro-tile (4 M-tiles sharing one
// B fragment per K-step => 5 b128-load-pairs per 4 WMMAs instead of 8).
// M must be a multiple of 64 (true here: 16384). 8 waves per block.
// ---------------------------------------------------------------------------
__global__ void wmma_gemm_kernel(const _Float16* __restrict__ A, const _Float16* __restrict__ B,
                                 float* __restrict__ C,
                                 const float* __restrict__ bias0, const float* __restrict__ bias1,
                                 int M, int N, int Kp, int accumulate) {
    int wave = threadIdx.x >> 5;
    int lane = threadIdx.x & 31;
    int tilesN = N >> 4;
    int macroM = M >> 6;                       // 64-row macro tiles
    int tile = blockIdx.x * (blockDim.x >> 5) + wave;
    int totalTiles = macroM * tilesN;
    if (tile >= totalTiles) return;            // uniform per wave -> EXEC all-1 inside WMMA
    int tmBase = (tile / tilesN) << 2;         // first of 4 consecutive 16-row tiles
    int tn     = tile % tilesN;

    v8f acc0 = {}, acc1 = {}, acc2 = {}, acc3 = {};
    for (int k0 = 0; k0 < Kp; k0 += 32) {
        v16h b  = load_frag16x32(B, Kp, tn * 16, k0);
        v16h a0 = load_frag16x32(A, Kp, (tmBase + 0) * 16, k0);
        v16h a1 = load_frag16x32(A, Kp, (tmBase + 1) * 16, k0);
        v16h a2 = load_frag16x32(A, Kp, (tmBase + 2) * 16, k0);
        v16h a3 = load_frag16x32(A, Kp, (tmBase + 3) * 16, k0);
        acc0 = __builtin_amdgcn_wmma_f32_16x16x32_f16(false, a0, false, b, (short)0, acc0, false, false);
        acc1 = __builtin_amdgcn_wmma_f32_16x16x32_f16(false, a1, false, b, (short)0, acc1, false, false);
        acc2 = __builtin_amdgcn_wmma_f32_16x16x32_f16(false, a2, false, b, (short)0, acc2, false, false);
        acc3 = __builtin_amdgcn_wmma_f32_16x16x32_f16(false, a3, false, b, (short)0, acc3, false, false);
    }

    int col = tn * 16 + (lane & 15);
    float bsum = 0.0f;
    if (bias0) bsum += bias0[col];
    if (bias1) bsum += bias1[col];

    v8f accs[4] = {acc0, acc1, acc2, acc3};
#pragma unroll
    for (int t = 0; t < 4; ++t) {
        int rbase = (tmBase + t) * 16 + ((lane & 16) ? 8 : 0);
#pragma unroll
        for (int i = 0; i < 8; ++i) {
            float* p = C + (size_t)(rbase + i) * N + col;
            float v = accs[t][i] + bsum + (accumulate ? *p : 0.0f);
            *p = v;
        }
    }
}

// ---------------------------------------------------------------------------
// Reset the inter-workgroup sync state for a recurrence phase.
// ---------------------------------------------------------------------------
__global__ void init_sync_kernel(unsigned* __restrict__ counter, float* __restrict__ hglob) {
    if (threadIdx.x == 0) *counter = 0u;
    if (threadIdx.x < 2 * HID) hglob[threadIdx.x] = 0.0f;
}

// ---------------------------------------------------------------------------
// Persistent LSTM recurrence. 10 workgroups, each owns 40 hidden units
// (all 4 gate rows => 160x400 fp32 weight slice resident in 256KB of LDS,
// possible only with CDNA5's 320KB-per-WGP LDS). Per step:
//   - load h_prev (400 floats) from L2-resident double buffer (agent-scope)
//   - 160 dot products, 2 threads/row, wave shuffle reduce
//   - gate nonlinearity, c/h update for the local 40 units
//   - publish h slice, global atomic barrier across the 10 WGs
// ---------------------------------------------------------------------------
__global__ void lstm_rec_kernel(const float* __restrict__ xg,   // [SEQ, 1600] pre-activations
                                const float* __restrict__ Whh,  // [1600, 400]
                                float* __restrict__ hseq,       // [SEQ, 400] output
                                float* __restrict__ hglob,      // [2*400] double buffer
                                unsigned* __restrict__ counter) {
    extern __shared__ float smem[];
    float* Wsl  = smem;                              // REC_ROWS * REC_WPITCH
    float* h_sh = Wsl + REC_ROWS * REC_WPITCH;       // 404
    float* pre  = h_sh + 404;                        // REC_ROWS
    float* cst  = pre + REC_ROWS;                    // REC_UPW

    const int U0 = blockIdx.x * REC_UPW;

    // Stage weight slice: local row r = gate*40 + unit <-> global row gate*400 + U0 + unit
    for (int idx = threadIdx.x; idx < REC_ROWS * HID; idx += REC_THREADS) {
        int r = idx / HID;
        int k = idx - r * HID;
        int gate = r / REC_UPW;
        int unit = r - gate * REC_UPW;
        Wsl[r * REC_WPITCH + k] = Whh[(size_t)(gate * HID + U0 + unit) * HID + k];
    }
    if (threadIdx.x < REC_UPW) cst[threadIdx.x] = 0.0f;
    __syncthreads();

    const int r   = threadIdx.x >> 1;   // 0..159
    const int hlf = threadIdx.x & 1;    // which half of the K=400 dot
    const float4* wrow = (const float4*)(Wsl + r * REC_WPITCH + hlf * 200);

    for (int s = 0; s < SEQ; ++s) {
        const float* hcur = hglob + ((s & 1) * HID);
        float*       hnxt = hglob + (((s & 1) ^ 1) * HID);

        for (int u = threadIdx.x; u < HID; u += REC_THREADS)
            h_sh[u] = __hip_atomic_load(&hcur[u], __ATOMIC_RELAXED, __HIP_MEMORY_SCOPE_AGENT);
        __syncthreads();

        const float4* hv = (const float4*)(h_sh + hlf * 200);
        float4 a = {0.0f, 0.0f, 0.0f, 0.0f};
#pragma unroll 10
        for (int q = 0; q < 50; ++q) {
            float4 w = wrow[q];
            float4 h = hv[q];
            a.x += w.x * h.x; a.y += w.y * h.y; a.z += w.z * h.z; a.w += w.w * h.w;
        }
        float sum = (a.x + a.y) + (a.z + a.w);
        sum += __shfl_xor(sum, 1, 32);
        if (hlf == 0) {
            int gate = r / REC_UPW;
            int unit = r - gate * REC_UPW;
            pre[r] = sum + xg[(size_t)s * G4H + gate * HID + U0 + unit];
        }
        __syncthreads();

        if (threadIdx.x < REC_UPW) {
            int u = threadIdx.x;
            float i_ = 1.0f / (1.0f + expf(-pre[u]));
            float f_ = 1.0f / (1.0f + expf(-pre[REC_UPW + u]));
            float g_ = tanhf(pre[2 * REC_UPW + u]);
            float o_ = 1.0f / (1.0f + expf(-pre[3 * REC_UPW + u]));
            float c  = f_ * cst[u] + i_ * g_;
            float hn = o_ * tanhf(c);
            cst[u] = c;
            hseq[(size_t)s * HID + U0 + u] = hn;
            __hip_atomic_store(&hnxt[U0 + u], hn, __ATOMIC_RELAXED, __HIP_MEMORY_SCOPE_AGENT);
        }
        __threadfence();
        __syncthreads();
        if (threadIdx.x == 0) {
            __hip_atomic_fetch_add(counter, 1u, __ATOMIC_ACQ_REL, __HIP_MEMORY_SCOPE_AGENT);
            unsigned target = (unsigned)REC_NWG * (unsigned)(s + 1);
            while (__hip_atomic_load(counter, __ATOMIC_ACQUIRE, __HIP_MEMORY_SCOPE_AGENT) < target)
                __builtin_amdgcn_s_sleep(2);
        }
        __syncthreads();
    }
}

// ---------------------------------------------------------------------------
// MDN head: y = lstm_out @ out_W.T + out_b per row, then nonlinearities.
// One block per sequence row (rows 0..16382). Output layout (flat fp32):
//   e[16383], pi[16383,20], mu1, mu2, sigma1, sigma2, rho
// ---------------------------------------------------------------------------
__global__ void out_head_kernel(const float* __restrict__ lstm_out,
                                const float* __restrict__ Wout,   // [121,400]
                                const float* __restrict__ bout,   // [121]
                                float* __restrict__ out) {
    __shared__ float row[HID];
    __shared__ float y[OUTC];
    int s = blockIdx.x;            // 0..16382
    for (int k = threadIdx.x; k < HID; k += blockDim.x)
        row[k] = lstm_out[(size_t)s * HID + k];
    __syncthreads();
    int c = threadIdx.x;
    if (c < OUTC) {
        const float* w = Wout + (size_t)c * HID;
        float acc = bout[c];
        for (int k = 0; k < HID; ++k) acc += w[k] * row[k];
        y[c] = acc;
    }
    __syncthreads();
    if (c < OUTC) {
        float v = y[c];
        if (c == 0) {
            out[s] = 1.0f / (1.0f + expf(-v));
        } else {
            int q = (c - 1) / 20;
            int j = (c - 1) - q * 20;
            float r;
            if (q == 0) {                       // softmax over pi columns
                float mx = -1e30f;
                for (int t = 0; t < 20; ++t) mx = fmaxf(mx, y[1 + t]);
                float sm = 0.0f;
                for (int t = 0; t < 20; ++t) sm += expf(y[1 + t] - mx);
                r = expf(v - mx) / sm;
            } else if (q == 1 || q == 2) {      // mu1, mu2
                r = v;
            } else if (q == 3 || q == 4) {      // sigma1, sigma2
                r = expf(v);
            } else {                            // rho
                r = tanhf(v);
            }
            out[(size_t)SM1 + (size_t)q * SM1 * 20 + (size_t)s * 20 + j] = r;
        }
    }
}

// ---------------------------------------------------------------------------
// Workspace layout (bytes, 256-aligned)
// ---------------------------------------------------------------------------
#define OFF_XG     ((size_t)0)                                 // SEQ*1600 f32 = 104857600
#define OFF_XSK1   (OFF_XG   + (size_t)SEQ * G4H * 4)          // SEQ*400 f32
#define OFF_XSK2   (OFF_XSK1 + (size_t)SEQ * HID * 4)
#define OFF_H1     (OFF_XSK2 + (size_t)SEQ * HID * 4)
#define OFF_H2     (OFF_H1   + (size_t)SEQ * HID * 4)
#define OFF_H3     (OFF_H2   + (size_t)SEQ * HID * 4)
#define OFF_LOUT   (OFF_H3   + (size_t)SEQ * HID * 4)
#define OFF_LIN16  (OFF_LOUT + (size_t)SEQ * HID * 4)          // SEQ*416 f16
#define OFF_W16    (OFF_LIN16 + (size_t)SEQ * KP * 2)          // 1600*416 f16
#define OFF_HGLOB  (OFF_W16  + (size_t)G4H * KP * 2)           // 2*400 f32 (padded)
#define OFF_CNT    (OFF_HGLOB + (size_t)3328)

extern "C" void kernel_launch(void* const* d_in, const int* in_sizes, int n_in,
                              void* d_out, int out_size, void* d_ws, size_t ws_size,
                              hipStream_t stream) {
    (void)in_sizes; (void)n_in; (void)out_size; (void)ws_size;

    const float* x      = (const float*)d_in[0];
    const float* l1_Wih = (const float*)d_in[1];
    const float* l1_Whh = (const float*)d_in[2];
    const float* l1_bih = (const float*)d_in[3];
    const float* l1_bhh = (const float*)d_in[4];
    const float* l2_Wih = (const float*)d_in[5];
    const float* l2_Whh = (const float*)d_in[6];
    const float* l2_bih = (const float*)d_in[7];
    const float* l2_bhh = (const float*)d_in[8];
    const float* l3_Wih = (const float*)d_in[9];
    const float* l3_Whh = (const float*)d_in[10];
    const float* l3_bih = (const float*)d_in[11];
    const float* l3_bhh = (const float*)d_in[12];
    const float* si1_W  = (const float*)d_in[13];
    const float* si1_b  = (const float*)d_in[14];
    const float* si2_W  = (const float*)d_in[15];
    const float* si2_b  = (const float*)d_in[16];
    const float* so1_W  = (const float*)d_in[17];
    const float* so1_b  = (const float*)d_in[18];
    const float* so2_W  = (const float*)d_in[19];
    const float* so2_b  = (const float*)d_in[20];
    const float* out_W  = (const float*)d_in[21];
    const float* out_b  = (const float*)d_in[22];

    char* ws = (char*)d_ws;
    float*     xg    = (float*)(ws + OFF_XG);
    float*     xsk1  = (float*)(ws + OFF_XSK1);
    float*     xsk2  = (float*)(ws + OFF_XSK2);
    float*     h1    = (float*)(ws + OFF_H1);
    float*     h2    = (float*)(ws + OFF_H2);
    float*     h3    = (float*)(ws + OFF_H3);
    float*     lout  = (float*)(ws + OFF_LOUT);
    _Float16*  lin16 = (_Float16*)(ws + OFF_LIN16);
    _Float16*  w16   = (_Float16*)(ws + OFF_W16);
    float*     hglob = (float*)(ws + OFF_HGLOB);
    unsigned*  cnt   = (unsigned*)(ws + OFF_CNT);

    hipFuncSetAttribute(reinterpret_cast<const void*>(lstm_rec_kernel),
                        hipFuncAttributeMaxDynamicSharedMemorySize, (int)REC_SMEM_BYTES);

    // macro-tiles: 64 rows x 16 cols per wave, 8 waves per block
    const int gemm_xg_blocks   = (((SEQ / 64) * (G4H / 16)) + 7) / 8;  // 3200
    const int gemm_skip_blocks = (((SEQ / 64) * (HID / 16)) + 7) / 8;  // 800
    const int cvt_lin_blocks   = (int)(((size_t)SEQ * KP + 255) / 256);
    const int cvt_wih_blocks   = (int)(((size_t)G4H * KP + 255) / 256);
    const int cvt_so_blocks    = (int)(((size_t)HID * KP + 255) / 256);

    // ---- input-side projections (independent of recurrence) ----
    proj3_kernel<<<(int)(((size_t)SEQ * G4H + 255) / 256), 256, 0, stream>>>(
        x, l1_Wih, l1_bih, l1_bhh, xg, G4H);
    proj3_kernel<<<(int)(((size_t)SEQ * HID + 255) / 256), 256, 0, stream>>>(
        x, si1_W, si1_b, nullptr, xsk1, HID);
    proj3_kernel<<<(int)(((size_t)SEQ * HID + 255) / 256), 256, 0, stream>>>(
        x, si2_W, si2_b, nullptr, xsk2, HID);

    // ---- layer 1 recurrence ----
    init_sync_kernel<<<1, 1024, 0, stream>>>(cnt, hglob);
    lstm_rec_kernel<<<REC_NWG, REC_THREADS, REC_SMEM_BYTES, stream>>>(xg, l1_Whh, h1, hglob, cnt);

    // ---- layer 2: xg = (h1 + xskip1) @ Wih2.T + b, then recurrence ----
    cvt_f16_pad_kernel<<<cvt_lin_blocks, 256, 0, stream>>>(h1, xsk1, lin16, SEQ, HID, KP);
    cvt_f16_pad_kernel<<<cvt_wih_blocks, 256, 0, stream>>>(l2_Wih, nullptr, w16, G4H, HID, KP);
    wmma_gemm_kernel<<<gemm_xg_blocks, 256, 0, stream>>>(lin16, w16, xg, l2_bih, l2_bhh,
                                                         SEQ, G4H, KP, 0);
    init_sync_kernel<<<1, 1024, 0, stream>>>(cnt, hglob);
    lstm_rec_kernel<<<REC_NWG, REC_THREADS, REC_SMEM_BYTES, stream>>>(xg, l2_Whh, h2, hglob, cnt);

    // ---- layer 3 ----
    cvt_f16_pad_kernel<<<cvt_lin_blocks, 256, 0, stream>>>(h2, xsk2, lin16, SEQ, HID, KP);
    cvt_f16_pad_kernel<<<cvt_wih_blocks, 256, 0, stream>>>(l3_Wih, nullptr, w16, G4H, HID, KP);
    wmma_gemm_kernel<<<gemm_xg_blocks, 256, 0, stream>>>(lin16, w16, xg, l3_bih, l3_bhh,
                                                         SEQ, G4H, KP, 0);
    init_sync_kernel<<<1, 1024, 0, stream>>>(cnt, hglob);
    lstm_rec_kernel<<<REC_NWG, REC_THREADS, REC_SMEM_BYTES, stream>>>(xg, l3_Whh, h3, hglob, cnt);

    // ---- lstm_out = h3 + h1@so1.T + b1 + h2@so2.T + b2 ----
    hipMemcpyAsync(lout, h3, (size_t)SEQ * HID * 4, hipMemcpyDeviceToDevice, stream);
    cvt_f16_pad_kernel<<<cvt_lin_blocks, 256, 0, stream>>>(h1, nullptr, lin16, SEQ, HID, KP);
    cvt_f16_pad_kernel<<<cvt_so_blocks, 256, 0, stream>>>(so1_W, nullptr, w16, HID, HID, KP);
    wmma_gemm_kernel<<<gemm_skip_blocks, 256, 0, stream>>>(lin16, w16, lout, so1_b, nullptr,
                                                           SEQ, HID, KP, 1);
    cvt_f16_pad_kernel<<<cvt_lin_blocks, 256, 0, stream>>>(h2, nullptr, lin16, SEQ, HID, KP);
    cvt_f16_pad_kernel<<<cvt_so_blocks, 256, 0, stream>>>(so2_W, nullptr, w16, HID, HID, KP);
    wmma_gemm_kernel<<<gemm_skip_blocks, 256, 0, stream>>>(lin16, w16, lout, so2_b, nullptr,
                                                           SEQ, HID, KP, 1);

    // ---- MDN head + nonlinearities ----
    out_head_kernel<<<SM1, 128, 0, stream>>>(lout, out_W, out_b, (float*)d_out);
}
